// QuantOPTDecoderLayer_28226525069369
// MI455X (gfx1250) — 8-bit matrix operand order repaired and checked by reading, compiled, not run on hardware
//
#include <hip/hip_runtime.h>
#include <hip/hip_bf16.h>

constexpr int S_  = 2048;
constexpr int D_  = 2048;
constexpr int H_  = 32;
constexpr int HD_ = 64;
constexpr int FF_ = 8192;
#define NEGF (-3.40282347e38f)

typedef __attribute__((ext_vector_type(16))) _Float16 v16h_t;
typedef __attribute__((ext_vector_type(8)))  float    v8f_t;
typedef __attribute__((ext_vector_type(8)))  int      v8i_t;

__device__ __forceinline__ v8f_t wmma_f16(v16h_t a, v16h_t b, v8f_t c) {
  return __builtin_amdgcn_wmma_f32_16x16x32_f16(false, a, false, b, (short)0, c, false, false);
}
__device__ __forceinline__ v8i_t wmma_u8(v8i_t a, v8i_t b, v8i_t c) {
  return __builtin_amdgcn_wmma_i32_16x16x64_iu8(false, a, false, b, c, false, false);
}

__device__ __forceinline__ void async_copy16(unsigned lds, const void* g) {
  asm volatile("global_load_async_to_lds_b128 %0, %1, off" :: "v"(lds), "v"(g) : "memory");
}
__device__ __forceinline__ void wait_async0() {
  asm volatile("s_wait_asynccnt 0x0" ::: "memory");
}

union F16Frag { v16h_t v; uint4 u[2]; };
__device__ __forceinline__ v16h_t load_f16_frag(const _Float16* p0, const _Float16* p1) {
  F16Frag f;
  f.u[0] = *(const uint4*)p0;
  f.u[1] = *(const uint4*)p1;
  return f.v;
}

__device__ __forceinline__ float block_reduce(float v, float* red, int op) {
  const int tid = threadIdx.x, n = blockDim.x;
  red[tid] = v; __syncthreads();
  for (int s = n >> 1; s > 0; s >>= 1) {
    if (tid < s) {
      float o = red[tid + s];
      red[tid] = (op == 0) ? (red[tid] + o) : ((op == 1) ? fmaxf(red[tid], o) : fminf(red[tid], o));
    }
    __syncthreads();
  }
  float r = red[0]; __syncthreads();
  return r;
}

__global__ __launch_bounds__(256) void ln_quant_u8_kernel(
    const float* __restrict__ x, const float* __restrict__ g, const float* __restrict__ b,
    unsigned char* __restrict__ q, float* __restrict__ s_out, float* __restrict__ zp_out,
    int* __restrict__ sum_out)
{
  const int C = D_;
  __shared__ float red[256];
  __shared__ float ybuf[D_];
  const int row = blockIdx.x, tid = threadIdx.x;
  const float* xr = x + (size_t)row * C;

  float s1 = 0.f, s2 = 0.f;
  for (int c = tid; c < C; c += 256) { float v = xr[c]; s1 += v; s2 += v * v; }
  s1 = block_reduce(s1, red, 0);
  s2 = block_reduce(s2, red, 0);
  const float mean = s1 / C;
  const float var  = s2 / C - mean * mean;
  const float rstd = rsqrtf(var + 1e-5f);

  float mn = 0.f, mx = 0.f;
  for (int c = tid; c < C; c += 256) {
    float y = (xr[c] - mean) * rstd * g[c] + b[c];
    ybuf[c] = y;
    mn = fminf(mn, y); mx = fmaxf(mx, y);
  }
  mn = block_reduce(mn, red, 2);
  mx = block_reduce(mx, red, 1);
  const float scale = fmaxf((mx - mn) * (1.f / 255.f), 1e-5f);
  const float zp    = rintf(-mn / scale);
  const float isc   = 1.f / scale;

  float qs = 0.f;
  for (int c = tid; c < C; c += 256) {
    float qv = fminf(fmaxf(rintf(ybuf[c] * isc) + zp, 0.f), 255.f);
    q[(size_t)row * C + c] = (unsigned char)qv;
    qs += qv;
  }
  qs = block_reduce(qs, red, 0);
  if (tid == 0) { s_out[row] = scale; zp_out[row] = zp; sum_out[row] = (int)qs; }
}

__global__ __launch_bounds__(256) void rowquant_u8_kernel(
    const float* __restrict__ x, unsigned char* __restrict__ q,
    float* __restrict__ s_out, float* __restrict__ zp_out, int* __restrict__ sum_out, int C)
{
  __shared__ float red[256];
  const int row = blockIdx.x, tid = threadIdx.x;
  const float* xr = x + (size_t)row * C;

  float mn = 0.f, mx = 0.f;
  for (int c = tid; c < C; c += 256) { float v = xr[c]; mn = fminf(mn, v); mx = fmaxf(mx, v); }
  mn = block_reduce(mn, red, 2);
  mx = block_reduce(mx, red, 1);
  const float scale = fmaxf((mx - mn) * (1.f / 255.f), 1e-5f);
  const float zp    = rintf(-mn / scale);
  const float isc   = 1.f / scale;

  float qs = 0.f;
  for (int c = tid; c < C; c += 256) {
    float qv = fminf(fmaxf(rintf(xr[c] * isc) + zp, 0.f), 255.f);
    q[(size_t)row * C + c] = (unsigned char)qv;
    qs += qv;
  }
  qs = block_reduce(qs, red, 0);
  if (tid == 0) { s_out[row] = scale; zp_out[row] = zp; sum_out[row] = (int)qs; }
}

__global__ __launch_bounds__(256) void rowquant_f16_kernel(
    const float* __restrict__ x, _Float16* __restrict__ dst, float alpha, int C)
{
  __shared__ float red[256];
  const int row = blockIdx.x, tid = threadIdx.x;
  const float* xr = x + (size_t)row * C;

  float mn = 0.f, mx = 0.f;
  for (int c = tid; c < C; c += 256) { float v = alpha * xr[c]; mn = fminf(mn, v); mx = fmaxf(mx, v); }
  mn = block_reduce(mn, red, 2);
  mx = block_reduce(mx, red, 1);
  const float scale = fmaxf((mx - mn) * (1.f / 255.f), 1e-5f);
  const float zp    = rintf(-mn / scale);
  const float isc   = 1.f / scale;

  for (int c = tid; c < C; c += 256) {
    float v  = alpha * xr[c];
    float qv = fminf(fmaxf(rintf(v * isc) + zp, 0.f), 255.f);
    dst[(size_t)row * C + c] = (_Float16)((qv - zp) * scale);
  }
}

__global__ __launch_bounds__(256) void transpose_f16_kernel(
    const _Float16* __restrict__ src, _Float16* __restrict__ dst, int R, int C)
{
  size_t idx = (size_t)blockIdx.x * 256 + threadIdx.x;
  int r = (int)(idx / C), c = (int)(idx % C);
  dst[(size_t)c * R + r] = src[idx];
}

__global__ __launch_bounds__(256) void gemm_iu8_kernel(
    const unsigned char* __restrict__ A, const unsigned char* __restrict__ W,
    const float* __restrict__ sx, const float* __restrict__ zpx, const int* __restrict__ sumx,
    const float* __restrict__ sw, const float* __restrict__ zpw, const int* __restrict__ sumw,
    const float* __restrict__ bias, const float* __restrict__ resid,
    float* __restrict__ out, int M, int N, int K, int mode)
{
  __shared__ __align__(16) unsigned char As[2][128][64];
  __shared__ __align__(16) unsigned char Bs[2][128][64];
  const int n0 = blockIdx.x * 128, m0 = blockIdx.y * 128;
  const int tid = threadIdx.x, w = tid >> 5, L = tid & 31;
  const int lm = L & 15, lh = L >> 4;
  const int r = tid >> 2, c16 = (tid & 3) * 16;

  const unsigned char* Ag0 = A + (size_t)(m0 + r)      * K + c16;
  const unsigned char* Ag1 = A + (size_t)(m0 + r + 64) * K + c16;
  const unsigned char* Wg0 = W + (size_t)(n0 + r)      * K + c16;
  const unsigned char* Wg1 = W + (size_t)(n0 + r + 64) * K + c16;

  v8i_t acc[8] = {};

  auto issue_tile = [&](int k0, int b) {
    async_copy16((unsigned)(uintptr_t)&As[b][r][c16],      Ag0 + k0);
    async_copy16((unsigned)(uintptr_t)&As[b][r + 64][c16], Ag1 + k0);
    async_copy16((unsigned)(uintptr_t)&Bs[b][r][c16],      Wg0 + k0);
    async_copy16((unsigned)(uintptr_t)&Bs[b][r + 64][c16], Wg1 + k0);
  };

  const int nT = K / 64;
  issue_tile(0, 0);
  int buf = 0;
  for (int t = 0; t < nT; ++t) {
    wait_async0();
    __syncthreads();
    if (t + 1 < nT) issue_tile((t + 1) * 64, buf ^ 1);

    v8i_t a;
    {
      const unsigned* Ar = (const unsigned*)&As[buf][w * 16 + lm][0];
      const int h2 = lh * 2;
#pragma unroll
      for (int cc = 0; cc < 4; cc++) { a[2 * cc] = (int)Ar[4 * cc + h2]; a[2 * cc + 1] = (int)Ar[4 * cc + h2 + 1]; }
    }
    const int kh4 = lh * 4;
#pragma unroll
    for (int ns = 0; ns < 8; ns++) {
      const unsigned* Br = (const unsigned*)&Bs[buf][ns * 16 + lm][0];
      v8i_t b;
#pragma unroll
      for (int j = 0; j < 4; j++) { b[2 * j] = (int)Br[4 * j + lh * 2]; b[2 * j + 1] = (int)Br[4 * j + lh * 2 + 1]; }
      acc[ns] = wmma_u8(a, b, acc[ns]);
    }
    __syncthreads();
    buf ^= 1;
  }

  const float Kf = (float)K;
  float zxm[8], sxm[8], smx[8];
#pragma unroll
  for (int v = 0; v < 8; v++) {
    int m = m0 + w * 16 + v + 8 * lh;
    zxm[v] = zpx[m]; sxm[v] = sx[m]; smx[v] = (float)sumx[m];
  }
#pragma unroll
  for (int ns = 0; ns < 8; ns++) {
    const int n = n0 + ns * 16 + lm;
    const float swn = sw[n], zwn = zpw[n], smw = (float)sumw[n], bn = bias[n];
#pragma unroll
    for (int v = 0; v < 8; v++) {
      const int m = m0 + w * 16 + v + 8 * lh;
      float rr = ((float)acc[ns][v] - zxm[v] * smw - zwn * smx[v] + Kf * zxm[v] * zwn) * sxm[v] * swn + bn;
      if (mode == 1)      rr += resid[(size_t)m * N + n];
      else if (mode == 2) rr  = fmaxf(rr, 0.f);
      out[(size_t)m * N + n] = rr;
    }
  }
}

__global__ __launch_bounds__(128) void attn_kernel(
    const _Float16* __restrict__ q16, const _Float16* __restrict__ k16,
    const _Float16* __restrict__ vT, float* __restrict__ ctx)
{
  extern __shared__ __align__(16) char smem[];
  float*    sc  = (float*)smem;
  _Float16* pb  = (_Float16*)(smem + (size_t)16 * S_ * 4);
  float*    red = (float*)(smem + (size_t)16 * S_ * 4 + (size_t)16 * S_ * 2);

  const int h = blockIdx.y, qt = blockIdx.x;
  const int tid = threadIdx.x, w = tid >> 5, L = tid & 31;
  const int lm = L & 15, lh8 = (L >> 4) * 8;

  const _Float16* qrow = q16 + (size_t)(qt * 16 + lm) * D_ + h * HD_;
  const v16h_t qa0 = load_f16_frag(qrow + lh8,      qrow + 16 + lh8);
  const v16h_t qa1 = load_f16_frag(qrow + 32 + lh8, qrow + 48 + lh8);
  for (int kt = w; kt < S_ / 16; kt += 4) {
    const _Float16* krow = k16 + (size_t)(kt * 16 + lm) * D_ + h * HD_;
    const v16h_t kb0 = load_f16_frag(krow + lh8,      krow + 16 + lh8);
    const v16h_t kb1 = load_f16_frag(krow + 32 + lh8, krow + 48 + lh8);
    v8f_t accs = {};
    accs = wmma_f16(qa0, kb0, accs);
    accs = wmma_f16(qa1, kb1, accs);
    const int j = kt * 16 + lm;
#pragma unroll
    for (int v = 0; v < 8; v++) {
      const int i = v + 8 * (L >> 4);
      float s = accs[v];
      if (j > qt * 16 + i) s = NEGF;
      sc[i * S_ + j] = s;
    }
  }
  __syncthreads();

  for (int rrow = 0; rrow < 16; rrow++) {
    float* srow = &sc[rrow * S_];
    float mx = NEGF;
    for (int c = tid; c < S_; c += 128) mx = fmaxf(mx, srow[c]);
    mx = block_reduce(mx, red, 1);
    float sum = 0.f;
    for (int c = tid; c < S_; c += 128) { float e = __expf(srow[c] - mx); srow[c] = e; sum += e; }
    sum = block_reduce(sum, red, 0);
    const float inv    = 1.f / sum;
    const float scale  = fmaxf(inv * (1.f / 255.f), 1e-5f);
    const float iscale = 1.f / scale;
    _Float16* prow = &pb[rrow * S_];
    for (int c = tid; c < S_; c += 128) {
      float p  = srow[c] * inv;
      float qv = fminf(fmaxf(rintf(p * iscale), 0.f), 255.f);
      prow[c]  = (_Float16)(qv * scale);
    }
    __syncthreads();
  }

  const _Float16* vrow = vT + (size_t)(h * HD_ + w * 16 + lm) * S_;
  v8f_t accc = {};
  for (int kk = 0; kk < S_ / 32; kk++) {
    const _Float16* pr = &pb[lm * S_ + kk * 32];
    const v16h_t a = load_f16_frag(pr + lh8, pr + 16 + lh8);
    const _Float16* vr = vrow + (size_t)kk * 32;
    const v16h_t b = load_f16_frag(vr + lh8, vr + 16 + lh8);
    accc = wmma_f16(a, b, accc);
  }
#pragma unroll
  for (int v = 0; v < 8; v++) {
    const int i = v + 8 * (L >> 4);
    ctx[(size_t)(qt * 16 + i) * D_ + h * HD_ + w * 16 + lm] = accc[v];
  }
}

extern "C" void kernel_launch(void* const* d_in, const int* in_sizes, int n_in,
                              void* d_out, int out_size, void* d_ws, size_t ws_size,
                              hipStream_t stream)
{
  (void)in_sizes; (void)n_in; (void)out_size; (void)ws_size;
  const float* hidden = (const float*)d_in[0];
  const float* Wqp = (const float*)d_in[2];  const float* bq = (const float*)d_in[3];
  const float* Wkp = (const float*)d_in[4];  const float* bk = (const float*)d_in[5];
  const float* Wvp = (const float*)d_in[6];  const float* bv = (const float*)d_in[7];
  const float* Wop = (const float*)d_in[8];  const float* bo = (const float*)d_in[9];
  const float* ln1g = (const float*)d_in[10]; const float* ln1b = (const float*)d_in[11];
  const float* W1p = (const float*)d_in[12]; const float* b1 = (const float*)d_in[13];
  const float* W2p = (const float*)d_in[14]; const float* b2 = (const float*)d_in[15];
  const float* ln2g = (const float*)d_in[16]; const float* ln2b = (const float*)d_in[17];
  float* out = (float*)d_out;

  size_t off = 0;
  char* base = (char*)d_ws;
  auto alloc = [&](size_t bytes) -> char* {
    char* p = base + off;
    off = (off + bytes + 255) & ~(size_t)255;
    return p;
  };

  unsigned char* aq   = (unsigned char*)alloc((size_t)S_ * FF_);
  float* a_s   = (float*)alloc((size_t)S_ * 4);
  float* a_zp  = (float*)alloc((size_t)S_ * 4);
  int*   a_sum = (int*)  alloc((size_t)S_ * 4);

  const int   wrows[6] = { D_, D_, D_, D_, FF_, D_ };
  const int   wcols[6] = { D_, D_, D_, D_, D_, FF_ };
  const float* Wsrc[6] = { Wqp, Wkp, Wvp, Wop, W1p, W2p };
  unsigned char* Wq8[6]; float* Ws[6]; float* Wz[6]; int* Wsum[6];
  for (int i = 0; i < 6; i++) {
    Wq8[i]  = (unsigned char*)alloc((size_t)wrows[i] * wcols[i]);
    Ws[i]   = (float*)alloc((size_t)wrows[i] * 4);
    Wz[i]   = (float*)alloc((size_t)wrows[i] * 4);
    Wsum[i] = (int*)  alloc((size_t)wrows[i] * 4);
  }

  _Float16* q16 = (_Float16*)alloc((size_t)S_ * D_ * 2);
  _Float16* k16 = (_Float16*)alloc((size_t)S_ * D_ * 2);
  _Float16* v16 = (_Float16*)alloc((size_t)S_ * D_ * 2);
  _Float16* vTh = (_Float16*)alloc((size_t)D_ * S_ * 2);
  float* tmp = (float*)alloc((size_t)S_ * FF_ * 4);
  float* h1  = (float*)alloc((size_t)S_ * D_ * 4);

  const dim3 gD(D_ / 128,  S_ / 128);
  const dim3 gF(FF_ / 128, S_ / 128);
  const size_t ATTN_SMEM = (size_t)16 * S_ * 4 + (size_t)16 * S_ * 2 + 128 * 4;

  for (int i = 0; i < 6; i++)
    rowquant_u8_kernel<<<wrows[i], 256, 0, stream>>>(Wsrc[i], Wq8[i], Ws[i], Wz[i], Wsum[i], wcols[i]);

  ln_quant_u8_kernel<<<S_, 256, 0, stream>>>(hidden, ln1g, ln1b, aq, a_s, a_zp, a_sum);

  gemm_iu8_kernel<<<gD, 256, 0, stream>>>(aq, Wq8[0], a_s, a_zp, a_sum, Ws[0], Wz[0], Wsum[0],
                                          bq, nullptr, tmp, S_, D_, D_, 0);
  rowquant_f16_kernel<<<S_, 256, 0, stream>>>(tmp, q16, 0.125f  , D_);
  gemm_iu8_kernel<<<gD, 256, 0, stream>>>(aq, Wq8[1], a_s, a_zp, a_sum, Ws[1], Wz[1], Wsum[1],
                                          bk, nullptr, tmp, S_, D_, D_, 0);
  rowquant_f16_kernel<<<S_, 256, 0, stream>>>(tmp, k16, 1.0f, D_);
  gemm_iu8_kernel<<<gD, 256, 0, stream>>>(aq, Wq8[2], a_s, a_zp, a_sum, Ws[2], Wz[2], Wsum[2],
                                          bv, nullptr, tmp, S_, D_, D_, 0);
  rowquant_f16_kernel<<<S_, 256, 0, stream>>>(tmp, v16, 1.0f, D_);
  transpose_f16_kernel<<<(S_ * D_) / 256, 256, 0, stream>>>(v16, vTh, S_, D_);

  attn_kernel<<<dim3(S_ / 16, H_), 128, ATTN_SMEM, stream>>>(q16, k16, vTh, tmp);

  rowquant_u8_kernel<<<S_, 256, 0, stream>>>(tmp, aq, a_s, a_zp, a_sum, D_);
  gemm_iu8_kernel<<<gD, 256, 0, stream>>>(aq, Wq8[3], a_s, a_zp, a_sum, Ws[3], Wz[3], Wsum[3],
                                          bo, hidden, h1, S_, D_, D_, 1);

  ln_quant_u8_kernel<<<S_, 256, 0, stream>>>(h1, ln2g, ln2b, aq, a_s, a_zp, a_sum);
  gemm_iu8_kernel<<<gF, 256, 0, stream>>>(aq, Wq8[4], a_s, a_zp, a_sum, Ws[4], Wz[4], Wsum[4],
                                          b1, nullptr, tmp, S_, FF_, D_, 2);
  rowquant_u8_kernel<<<S_, 256, 0, stream>>>(tmp, aq, a_s, a_zp, a_sum, FF_);
  gemm_iu8_kernel<<<gD, 256, 0, stream>>>(aq, Wq8[5], a_s, a_zp, a_sum, Ws[5], Wz[5], Wsum[5],
                                          b2, h1, out, S_, D_, FF_, 1);
}
